// CrossAttBlock_43035572306650
// MI455X (gfx1250) — matrix operand order repaired and checked by reading, not compiled, not run on hardware
//
#include <hip/hip_runtime.h>

// ---------------- problem constants ----------------
#define D_MODEL 1024
#define FF_DIM  4096
#define NHEAD   16
#define HD      64
#define BATCH   4
#define SEQ     1024
#define TOK     (BATCH*SEQ) // 4096 tokens

// ---------------- CDNA5 WMMA / TDM types ----------------
typedef __attribute__((ext_vector_type(16))) __bf16 v16bf;
typedef __attribute__((ext_vector_type(8)))  __bf16 v8bf;
typedef __attribute__((ext_vector_type(4)))  __bf16 v4bf;
typedef __attribute__((ext_vector_type(8)))  float  v8f;
typedef __attribute__((ext_vector_type(4)))  unsigned int u32x4;
typedef __attribute__((ext_vector_type(8)))  int         i32x8;
typedef __attribute__((ext_vector_type(4)))  int         i32x4;

union V16 { v16bf v; v8bf h[2]; };

__device__ __forceinline__ __bf16 to_bf16(float f) { return (__bf16)f; }

__device__ __forceinline__ v8f zero8() {
  v8f v;
#pragma unroll
  for (int i = 0; i < 8; ++i) v[i] = 0.0f;
  return v;
}

// =====================================================================
// TDM: DMA a 2D bf16 tile (tile_rows x 32 elems) from global into LDS,
// with TDM padding giving an LDS row pitch of 40 elements (32 data +
// 4-DWORD pad every 16 DWORDs).  Issued by one wave; tracked on TENSORcnt.
// =====================================================================
__device__ __forceinline__ void tdm_load_tile(unsigned lds_addr, const __bf16* gptr,
                                              unsigned tile_rows, unsigned row_stride_elems) {
  const unsigned long long ga = (unsigned long long)(size_t)gptr;
  u32x4 g0;
  g0[0] = 1u;                                           // count=1, user descriptor
  g0[1] = lds_addr;                                     // LDS byte address
  g0[2] = (unsigned)(ga & 0xFFFFFFFFu);                 // global addr lo
  g0[3] = (unsigned)((ga >> 32) & 0x1FFFFFFu)           // global addr hi (bits 56:32)
        | (2u << 30);                                   // type = 2 (image)
  const unsigned td0 = row_stride_elems;                // tensor dim0 (elements)
  const unsigned td1 = 1u << 20;                        // tensor dim1 (large, no clip)
  const unsigned long long st0 = row_stride_elems;      // dim0 stride (elements)
  i32x8 g1;
  g1[0] = (int)((1u << 16)                              // data_size = 1 (2 bytes)
              | (1u << 20)                              // pad_enable
              | (3u << 22)                              // pad_interval: 16 DWORDs
              | (3u << 25));                            // pad_amount: 4 DWORDs
  g1[1] = (int)((td0 & 0xFFFFu) << 16);                 // [47:32]=atomic addr 0, [63:48]=td0.lo
  g1[2] = (int)(((td0 >> 16) & 0xFFFFu) | ((td1 & 0xFFFFu) << 16));
  g1[3] = (int)(((td1 >> 16) & 0xFFFFu) | (32u << 16)); // tile_dim0 = 32 elems
  g1[4] = (int)(tile_rows & 0xFFFFu);                   // tile_dim1; tile_dim2 = 0
  g1[5] = (int)(st0 & 0xFFFFFFFFu);
  g1[6] = (int)((st0 >> 32) & 0xFFFFu);                 // stride hi; dim1_stride lo = 0
  g1[7] = 0;
  const i32x4 z = {0, 0, 0, 0};
#if __clang_major__ >= 23
  const i32x8 z8 = {0, 0, 0, 0, 0, 0, 0, 0};
  __builtin_amdgcn_tensor_load_to_lds(g0, g1, z, z, z8, 0);
#else
  __builtin_amdgcn_tensor_load_to_lds(g0, g1, z, z, 0);
#endif
}

// =====================================================================
// Prep kernels: one-time f32 -> bf16 conversions
// =====================================================================
// W[K,N] f32 -> Wt[N,K] bf16 (transposed). grid(N/32, K/32), 256 thr.
__global__ __launch_bounds__(256)
void convert_transpose(const float* __restrict__ W, __bf16* __restrict__ Wt,
                       int K, int N) {
  __shared__ float t[32][33];
  const int k0 = blockIdx.y * 32, n0 = blockIdx.x * 32;
  const int tx = threadIdx.x & 31, ty = threadIdx.x >> 5;
#pragma unroll
  for (int i = 0; i < 4; ++i)
    t[ty + i * 8][tx] = W[(size_t)(k0 + ty + i * 8) * N + n0 + tx];
  __syncthreads();
#pragma unroll
  for (int i = 0; i < 4; ++i)
    Wt[(size_t)(n0 + ty + i * 8) * K + k0 + tx] = to_bf16(t[tx][ty + i * 8]);
}

// elementwise f32 -> bf16, n multiple of 1024. grid(n/1024), 256 thr.
__global__ __launch_bounds__(256)
void convert_bf16(const float* __restrict__ X, __bf16* __restrict__ Y) {
  const size_t i = ((size_t)blockIdx.x * 256 + threadIdx.x) * 4;
  const float4 v = *(const float4*)(X + i);
  v4bf b;
  b[0] = to_bf16(v.x); b[1] = to_bf16(v.y);
  b[2] = to_bf16(v.z); b[3] = to_bf16(v.w);
  *(v4bf*)(Y + i) = b;
}

// =====================================================================
// GEMM: C[M,N] = epi(A[M,K] @ Wt[N,K]^T + bias) (+res).  A,Wt bf16.
// TDM double-buffered LDS tiles; 256 thr (8 waves), block 128x128,
// wave tile 32x64, K-step 32. Output bf16 (Cb) or f32 (Cf, +res).
// =====================================================================
template <bool RELU, bool BF16OUT>
__global__ __launch_bounds__(256)
void gemm_wmma_tdm(const __bf16* __restrict__ A, const __bf16* __restrict__ Wt,
                   const float* __restrict__ bias, const float* res,
                   float* Cf, __bf16* Cb, int M, int N, int K) {
  constexpr int BM = 128, BN = 128, BK = 32, ST = 40;  // ST = TDM-padded pitch
  __shared__ __bf16 As[2][BM * ST];
  __shared__ __bf16 Bs[2][BN * ST];

  const int tid = threadIdx.x;
  const int lane = tid & 31, wave = tid >> 5;
  const int wm = wave & 3, wn = wave >> 2;
  const int l16 = lane & 15, hi = lane >> 4;
  const int m0 = blockIdx.y * BM, n0 = blockIdx.x * BN;

  const __bf16* Ab = A  + (size_t)m0 * K;
  const __bf16* Bb = Wt + (size_t)n0 * K;

  v8f acc[2][4];
#pragma unroll
  for (int i = 0; i < 2; ++i)
#pragma unroll
    for (int j = 0; j < 4; ++j) acc[i][j] = zero8();

  if (tid < 32) {  // wave 0 drives the TDM
    tdm_load_tile((unsigned)(size_t)&As[0][0], Ab, BM, (unsigned)K);
    tdm_load_tile((unsigned)(size_t)&Bs[0][0], Bb, BN, (unsigned)K);
  }

  int buf = 0;
  for (int k0 = 0; k0 < K; k0 += BK) {
    if (tid < 32) __builtin_amdgcn_s_wait_tensorcnt(0);  // tiles for k0 resident
    __syncthreads();                                     // ...visible to all; prev reads done
    if (tid < 32 && k0 + BK < K) {                       // prefetch next pair (overlaps WMMA)
      tdm_load_tile((unsigned)(size_t)&As[buf ^ 1][0], Ab + k0 + BK, BM, (unsigned)K);
      tdm_load_tile((unsigned)(size_t)&Bs[buf ^ 1][0], Bb + k0 + BK, BN, (unsigned)K);
    }

    // hoisted fragment loads: 12 ds_load_b128 feeding 8 WMMAs
    const int kb = hi * 8;
    V16 a[2], b[4];
#pragma unroll
    for (int mi = 0; mi < 2; ++mi) {
      const int ar = wm * 32 + mi * 16 + l16;            // A: row = lane%16
      a[mi].h[0] = *(const v8bf*)(&As[buf][ar * ST + kb]);
      a[mi].h[1] = *(const v8bf*)(&As[buf][ar * ST + 16 + kb]);
    }
#pragma unroll
    for (int ni = 0; ni < 4; ++ni) {
      const int bc = wn * 64 + ni * 16 + l16;            // B: col = lane%16
      b[ni].h[0] = *(const v8bf*)(&Bs[buf][bc * ST + kb]);
      b[ni].h[1] = *(const v8bf*)(&Bs[buf][bc * ST + 16 + kb]);
    }
#pragma unroll
    for (int mi = 0; mi < 2; ++mi)
#pragma unroll
      for (int ni = 0; ni < 4; ++ni)
        acc[mi][ni] = __builtin_amdgcn_wmma_f32_16x16x32_bf16(
            false, a[mi].v, false, b[ni].v, (short)0, acc[mi][ni], false, false);
    buf ^= 1;
  }

#pragma unroll
  for (int mi = 0; mi < 2; ++mi) {
#pragma unroll
    for (int ni = 0; ni < 4; ++ni) {
      const int n = n0 + wn * 64 + ni * 16 + l16;
      const float bsv = bias ? bias[n] : 0.0f;
#pragma unroll
      for (int r = 0; r < 8; ++r) {
        const int m = m0 + wm * 32 + mi * 16 + r + hi * 8;
        float v = acc[mi][ni][r] + bsv;
        if (RELU) v = v > 0.0f ? v : 0.0f;
        if (BF16OUT) {
          Cb[(size_t)m * N + n] = to_bf16(v);
        } else {
          if (res) v += res[(size_t)m * N + n];
          Cf[(size_t)m * N + n] = v;
        }
      }
    }
  }
}

// =====================================================================
// LayerNorm: f32 in -> bf16 out. One 256-thread block per row.
// =====================================================================
__global__ __launch_bounds__(256)
void layernorm_k(const float* __restrict__ X, const float* __restrict__ g,
                 const float* __restrict__ b, __bf16* __restrict__ Y) {
  const int row = blockIdx.x;
  const float* x = X + (size_t)row * D_MODEL;
  float s = 0.0f, s2 = 0.0f;
  for (int i = threadIdx.x; i < D_MODEL; i += 256) {
    const float v = x[i];
    s += v; s2 += v * v;
  }
#pragma unroll
  for (int m = 16; m >= 1; m >>= 1) {
    s  += __shfl_xor(s,  m, 32);
    s2 += __shfl_xor(s2, m, 32);
  }
  __shared__ float rs[8], rs2[8];
  if ((threadIdx.x & 31) == 0) { rs[threadIdx.x >> 5] = s; rs2[threadIdx.x >> 5] = s2; }
  __syncthreads();
  float ts = 0.0f, ts2 = 0.0f;
#pragma unroll
  for (int i = 0; i < 8; ++i) { ts += rs[i]; ts2 += rs2[i]; }
  const float mu   = ts * (1.0f / D_MODEL);
  const float var  = ts2 * (1.0f / D_MODEL) - mu * mu;
  const float rstd = rsqrtf(var + 1e-5f);
  for (int i = threadIdx.x; i < D_MODEL; i += 256)
    Y[(size_t)row * D_MODEL + i] = to_bf16((x[i] - mu) * rstd * g[i] + b[i]);
}

// =====================================================================
// Flash attention, bf16 in / bf16 out.  Grid: (SEQ/64, BATCH*NHEAD),
// 128 thr = 4 waves, each wave owns 16 query rows; 1/sqrt(64) applied
// post-WMMA in f32.
// =====================================================================
__global__ __launch_bounds__(128)
void flash_attn(const __bf16* __restrict__ Q, const __bf16* __restrict__ Kv,
                const __bf16* __restrict__ Vv, __bf16* __restrict__ O) {
  const int qblk = blockIdx.x, bh = blockIdx.y;
  const int b = bh / NHEAD, h = bh % NHEAD;
  const int tid = threadIdx.x;
  const int lane = tid & 31, wave = tid >> 5;
  const int l16 = lane & 15, hi = lane >> 4;

  const __bf16* Qb = Q  + (size_t)b * SEQ * D_MODEL + h * HD;
  const __bf16* Kb = Kv + (size_t)b * SEQ * D_MODEL + h * HD;
  const __bf16* Vb = Vv + (size_t)b * SEQ * D_MODEL + h * HD;
  __bf16*       Ob = O  + (size_t)b * SEQ * D_MODEL + h * HD;

  const int q0 = qblk * 64 + wave * 16;
  const int kb = hi * 8;

  // Q fragments: raw 16B bf16 loads, no conversion
  V16 qf[2];
  {
    const __bf16* qp = Qb + (size_t)(q0 + l16) * D_MODEL;
#pragma unroll
    for (int c = 0; c < 2; ++c) {
      qf[c].h[0] = *(const v8bf*)(qp + c * 32 + kb);
      qf[c].h[1] = *(const v8bf*)(qp + c * 32 + 16 + kb);
    }
  }

  __shared__ __bf16 Pst[4][16][72];   // per-wave P staging (C -> A relayout)
  __shared__ __bf16 Vt[64][72];       // V tile transposed [dim][key]

  float mrow[8], lrow[8];
  v8f accO[4];
#pragma unroll
  for (int r = 0; r < 8; ++r) { mrow[r] = -1e30f; lrow[r] = 0.0f; }
#pragma unroll
  for (int t = 0; t < 4; ++t) accO[t] = zero8();

  for (int j = 0; j < SEQ / 64; ++j) {
    __syncthreads();
    // stage V tile (64 keys x 64 dims) transposed, bf16 -> bf16
#pragma unroll
    for (int i = 0; i < 4; ++i) {
      const int e8 = tid + i * 128;      // 0..511
      const int r  = e8 >> 3;            // key 0..63
      const int c  = (e8 & 7) * 8;       // dim start
      const v8bf v = *(const v8bf*)(Vb + (size_t)(j * 64 + r) * D_MODEL + c);
#pragma unroll
      for (int t = 0; t < 8; ++t) Vt[c + t][r] = v[t];
    }
    __syncthreads();

    // S tile = Q K^T (K fragments straight from global)
    v8f accS[4];
#pragma unroll
    for (int t = 0; t < 4; ++t) accS[t] = zero8();
#pragma unroll
    for (int nt = 0; nt < 4; ++nt) {
      const int krow = j * 64 + nt * 16 + l16;
#pragma unroll
      for (int c = 0; c < 2; ++c) {
        const __bf16* kp = Kb + (size_t)krow * D_MODEL + c * 32 + kb;
        V16 kf;
        kf.h[0] = *(const v8bf*)(kp);
        kf.h[1] = *(const v8bf*)(kp + 16);
        accS[nt] = __builtin_amdgcn_wmma_f32_16x16x32_bf16(
            false, qf[c].v, false, kf.v, (short)0, accS[nt], false, false);
      }
    }
#pragma unroll
    for (int t = 0; t < 4; ++t)
#pragma unroll
      for (int r = 0; r < 8; ++r) accS[t][r] *= 0.125f;   // 1/sqrt(64)

    // online softmax in C-register layout (rows M = r + 8*hi)
    float pv[4][8];
#pragma unroll
    for (int r = 0; r < 8; ++r) {
      float mx = fmaxf(fmaxf(accS[0][r], accS[1][r]), fmaxf(accS[2][r], accS[3][r]));
#pragma unroll
      for (int m = 1; m <= 8; m <<= 1) mx = fmaxf(mx, __shfl_xor(mx, m, 32));
      const float mnew = fmaxf(mrow[r], mx);
      const float corr = __expf(mrow[r] - mnew);
      mrow[r] = mnew;
      float rsum = 0.0f;
#pragma unroll
      for (int nt = 0; nt < 4; ++nt) {
        const float p = __expf(accS[nt][r] - mnew);
        pv[nt][r] = p; rsum += p;
      }
#pragma unroll
      for (int m = 1; m <= 8; m <<= 1) rsum += __shfl_xor(rsum, m, 32);
      lrow[r] = lrow[r] * corr + rsum;
#pragma unroll
      for (int nt = 0; nt < 4; ++nt) accO[nt][r] *= corr;
    }

    // P: C-layout -> LDS -> A-fragments
#pragma unroll
    for (int nt = 0; nt < 4; ++nt)
#pragma unroll
      for (int r = 0; r < 8; ++r)
        Pst[wave][r + hi * 8][nt * 16 + l16] = to_bf16(pv[nt][r]);

    // O += P @ V
#pragma unroll
    for (int c = 0; c < 2; ++c) {
      V16 pf;
      pf.h[0] = *(const v8bf*)(&Pst[wave][l16][c * 32 + kb]);
      pf.h[1] = *(const v8bf*)(&Pst[wave][l16][c * 32 + 16 + kb]);
#pragma unroll
      for (int nt = 0; nt < 4; ++nt) {
        V16 vf;
        vf.h[0] = *(const v8bf*)(&Vt[nt * 16 + l16][c * 32 + kb]);
        vf.h[1] = *(const v8bf*)(&Vt[nt * 16 + l16][c * 32 + 16 + kb]);
        accO[nt] = __builtin_amdgcn_wmma_f32_16x16x32_bf16(
            false, pf.v, false, vf.v, (short)0, accO[nt], false, false);
      }
    }
  }

#pragma unroll
  for (int nt = 0; nt < 4; ++nt)
#pragma unroll
    for (int r = 0; r < 8; ++r) {
      const int m = q0 + r + hi * 8;
      Ob[(size_t)m * D_MODEL + nt * 16 + l16] = to_bf16(accO[nt][r] / lrow[r]);
    }
}

// =====================================================================
// Host-side orchestration
// =====================================================================
extern "C" void kernel_launch(void* const* d_in, const int* in_sizes, int n_in,
                              void* d_out, int out_size, void* d_ws, size_t ws_size,
                              hipStream_t stream) {
  (void)in_sizes; (void)n_in; (void)out_size; (void)ws_size;
  auto F = [&](int i) { return (const float*)d_in[i]; };

  // Input index map (setup_inputs dict order, depth-first):
  // attnh=0..7, attn_hc=8..15, attnl=16..23, attn_lc=24..31  (wq,bq,wk,bk,wv,bv,wo,bo)
  // ffh=32..35, ffl=36..39 (w1,b1,w2,b2)
  // normh1=40,41 normh2=42,43 norml1=44,45 norml2=46,47 (g,b)
  // h=48, l=49
  const int AB[4] = {0, 8, 16, 24};   // attnh, attn_hc, attnl, attn_lc
  const float* h = F(48);
  const float* l = F(49);
  float* out_h = (float*)d_out;
  float* out_l = out_h + (size_t)TOK * D_MODEL;

  // ---- workspace bump allocator (bf16 regions) ----
  char* wp = (char*)d_ws;
  auto alloc = [&](size_t elems) { __bf16* r = (__bf16*)wp; wp += elems * 2; return r; };
  const size_t DD = (size_t)D_MODEL * D_MODEL;       // 1M
  const size_t DF = (size_t)D_MODEL * FF_DIM;        // 4M
  const size_t TD = (size_t)TOK * D_MODEL;           // 4M

  __bf16* wtq[4]; __bf16* wtk[4]; __bf16* wtv[4]; __bf16* wto[4];
  for (int a = 0; a < 4; ++a) {
    wtq[a] = alloc(DD); wtk[a] = alloc(DD); wto[a] = alloc(DD);
    wtv[a] = (a == 0 || a == 2) ? alloc(DD) : nullptr;  // self-attn only
  }
  __bf16* wt1[2]; __bf16* wt2[2];
  wt1[0] = alloc(DF); wt2[0] = alloc(DF);
  wt1[1] = alloc(DF); wt2[1] = alloc(DF);
  __bf16* hb  = alloc(TD);
  __bf16* lb  = alloc(TD);
  __bf16* Qb  = alloc(TD);
  __bf16* Kb  = alloc(TD);
  __bf16* Vb  = alloc(TD);
  __bf16* AO  = alloc(TD);
  __bf16* LNb = alloc(TD);
  __bf16* HID = alloc((size_t)TOK * FF_DIM);

  const dim3 thr(256);
  const dim3 athr(128);
  const dim3 agrid(SEQ / 64, BATCH * NHEAD);
  const dim3 lngrid(TOK);

  auto trans = [&](const float* W, __bf16* Wt, int K, int N) {
    convert_transpose<<<dim3(N / 32, K / 32), thr, 0, stream>>>(W, Wt, K, N);
  };
  auto gemmB = [&](const __bf16* A, const __bf16* Wt, const float* bias,
                   __bf16* C, int N, int K, bool relu) {
    dim3 grid(N / 128, TOK / 128);
    if (relu)
      gemm_wmma_tdm<true, true><<<grid, thr, 0, stream>>>(A, Wt, bias, nullptr, nullptr, C, TOK, N, K);
    else
      gemm_wmma_tdm<false, true><<<grid, thr, 0, stream>>>(A, Wt, bias, nullptr, nullptr, C, TOK, N, K);
  };
  auto gemmF = [&](const __bf16* A, const __bf16* Wt, const float* bias,
                   const float* res, float* C, int N, int K) {
    dim3 grid(N / 128, TOK / 128);
    gemm_wmma_tdm<false, false><<<grid, thr, 0, stream>>>(A, Wt, bias, res, C, nullptr, TOK, N, K);
  };

  // ---- prep: weights -> bf16 transposed, activations -> bf16 ----
  for (int a = 0; a < 4; ++a) {
    trans(F(AB[a] + 0), wtq[a], D_MODEL, D_MODEL);
    trans(F(AB[a] + 2), wtk[a], D_MODEL, D_MODEL);
    trans(F(AB[a] + 6), wto[a], D_MODEL, D_MODEL);
    if (wtv[a]) trans(F(AB[a] + 4), wtv[a], D_MODEL, D_MODEL);
  }
  trans(F(32), wt1[0], D_MODEL, FF_DIM);  // ffh.w1 [1024,4096]
  trans(F(34), wt2[0], FF_DIM, D_MODEL);  // ffh.w2 [4096,1024]
  trans(F(36), wt1[1], D_MODEL, FF_DIM);
  trans(F(38), wt2[1], FF_DIM, D_MODEL);
  convert_bf16<<<dim3(TD / 1024), thr, 0, stream>>>(h, hb);
  convert_bf16<<<dim3(TD / 1024), thr, 0, stream>>>(l, lb);

  // ---- self attention on h: at_h = h + attn(attnh, h) ----
  gemmB(hb, wtq[0], F(1), Qb, D_MODEL, D_MODEL, false);
  gemmB(hb, wtk[0], F(3), Kb, D_MODEL, D_MODEL, false);
  gemmB(hb, wtv[0], F(5), Vb, D_MODEL, D_MODEL, false);
  flash_attn<<<agrid, athr, 0, stream>>>(Qb, Kb, Vb, AO);
  gemmF(AO, wto[0], F(7), h, out_h, D_MODEL, D_MODEL);

  // ---- self attention on l: at_l = l + attn(attnl, l) ----
  gemmB(lb, wtq[2], F(17), Qb, D_MODEL, D_MODEL, false);
  gemmB(lb, wtk[2], F(19), Kb, D_MODEL, D_MODEL, false);
  gemmB(lb, wtv[2], F(21), Vb, D_MODEL, D_MODEL, false);
  flash_attn<<<agrid, athr, 0, stream>>>(Qb, Kb, Vb, AO);
  gemmF(AO, wto[2], F(23), l, out_l, D_MODEL, D_MODEL);

  // ---- cross attention h <- l : at_h += attn(attn_hc, LN(h), l); V = K (ref bug) ----
  layernorm_k<<<lngrid, thr, 0, stream>>>(h, F(40), F(41), LNb);
  gemmB(LNb, wtq[1], F(9),  Qb, D_MODEL, D_MODEL, false);
  gemmB(lb,  wtk[1], F(11), Kb, D_MODEL, D_MODEL, false);
  flash_attn<<<agrid, athr, 0, stream>>>(Qb, Kb, Kb, AO);
  gemmF(AO, wto[1], F(15), out_h, out_h, D_MODEL, D_MODEL);

  // ---- cross attention l <- h : at_l += attn(attn_lc, LN(l), h); V = K (ref bug) ----
  layernorm_k<<<lngrid, thr, 0, stream>>>(l, F(44), F(45), LNb);
  gemmB(LNb, wtq[3], F(25), Qb, D_MODEL, D_MODEL, false);
  gemmB(hb,  wtk[3], F(27), Kb, D_MODEL, D_MODEL, false);
  flash_attn<<<agrid, athr, 0, stream>>>(Qb, Kb, Kb, AO);
  gemmF(AO, wto[3], F(31), out_l, out_l, D_MODEL, D_MODEL);

  // ---- FF on h: h_out = at_h + relu(LN(at_h) @ w1 + b1) @ w2 + b2 ----
  layernorm_k<<<lngrid, thr, 0, stream>>>(out_h, F(42), F(43), LNb);
  gemmB(LNb, wt1[0], F(33), HID, FF_DIM, D_MODEL, true);
  gemmF(HID, wt2[0], F(35), out_h, out_h, D_MODEL, FF_DIM);

  // ---- FF on l ----
  layernorm_k<<<lngrid, thr, 0, stream>>>(out_l, F(46), F(47), LNb);
  gemmB(LNb, wt1[1], F(37), HID, FF_DIM, D_MODEL, true);
  gemmF(HID, wt2[1], F(39), out_l, out_l, D_MODEL, FF_DIM);
}
